// GMM_6176162971962
// MI455X (gfx1250) — matrix operand order repaired and checked by reading, not compiled, not run on hardware
//
#include <hip/hip_runtime.h>
#include <hip/hip_bf16.h>

typedef __attribute__((ext_vector_type(16))) __bf16 v16bf;
typedef __attribute__((ext_vector_type(8)))  __bf16 v8bf;
typedef __attribute__((ext_vector_type(8)))  float  v8f;
typedef __attribute__((ext_vector_type(4)))  unsigned int u32x4;
typedef __attribute__((ext_vector_type(8)))  int i32x8;
typedef __attribute__((ext_vector_type(4)))  int i32x4;

// GEMM tiling constants
#define TILE_M   128      // block tile M (8 waves x 16)
#define TILE_N   64       // block tile N
#define KC       64       // K chunk staged in LDS per TDM transfer
#define BPAD     72       // bf16 per LDS row after TDM padding (32 dw + 4 dw pad = 36 dw)

// ---------------------------------------------------------------------------
// TDM: DMA a 2D tile (KC x TILE_N rows) of bf16 from global (row stride
// ld_elems) into LDS at lds_off, inserting 4 DWORDs of padding after every
// 32 DWORDs (one 64-elem row) so LDS rows are 36 DWORDs => the 16-lane
// ds_load_b128 fragment pattern is bank-conflict-free (36n mod 64 distinct).
// Descriptor bitfields per CDNA5 ISA.
// ---------------------------------------------------------------------------
static __device__ __forceinline__ void tdm_load_tile(unsigned lds_off,
                                                     const __bf16* gptr,
                                                     unsigned ld_elems,
                                                     unsigned rows_total) {
    const unsigned long long ga = (unsigned long long)(const void*)gptr;
    u32x4 g0;
    g0[0] = 1u;                                     // count=1 valid, user D#
    g0[1] = lds_off;                                // lds_addr (bytes)
    g0[2] = (unsigned)(ga & 0xffffffffu);           // global_addr[31:0]
    g0[3] = (unsigned)((ga >> 32) & 0x01ffffffu)    // global_addr[56:32]
          | (2u << 30);                             // type = 2 (image)

    i32x8 g1;
    const unsigned data_size    = 1u;               // 2-byte elements
    const unsigned pad_enable   = 1u;
    const unsigned pad_interval = 4u;               // 2^(4+1) = 32 DWORDs
    const unsigned pad_amount   = 3u;               // 3+1 = 4 DWORDs
    g1[0] = (int)((data_size << 16) | (pad_enable << 20) |
                  (pad_interval << 22) | (pad_amount << 25));
    const unsigned td0 = ld_elems;                  // tensor_dim0 (K extent)
    const unsigned td1 = rows_total;                // tensor_dim1 (N rows)
    g1[1] = (int)((td0 & 0xffffu) << 16);                           // bits 63:48
    g1[2] = (int)(((td0 >> 16) & 0xffffu) | ((td1 & 0xffffu) << 16));
    g1[3] = (int)(((td1 >> 16) & 0xffffu) | ((unsigned)KC << 16));  // tile_dim0
    g1[4] = (int)((unsigned)TILE_N);                // tile_dim1; tile_dim2 = 0
    const unsigned long long s0 = ld_elems;         // tensor_dim0_stride
    g1[5] = (int)(s0 & 0xffffffffu);
    g1[6] = (int)((s0 >> 32) & 0xffffu);            // dim1_stride = 0
    g1[7] = 0;

    const i32x4 z4 = {0, 0, 0, 0};
#if __clang_major__ >= 23
    const i32x8 z8 = {0, 0, 0, 0, 0, 0, 0, 0};
    __builtin_amdgcn_tensor_load_to_lds(g0, g1, z4, z4, z8, 0);
#else
    __builtin_amdgcn_tensor_load_to_lds(g0, g1, z4, z4, 0);
#endif
}

// ---------------------------------------------------------------------------
// WMMA fragment helpers (v_wmma_f32_16x16x32_bf16, wave32)
// A (16x32): lane m = lane&15, koff = (lane&16)?8:0; chunks K=koff..+7 and
//            K=koff+16..+23 (two contiguous 16B loads).
// B (32x16): lane n = lane&15; K=koff..+7 and K=koff+16..+23, koff=(lane&16)?8:0
//            -> read from the LDS-resident B^T tile (row-major, BPAD-element rows).
// ---------------------------------------------------------------------------
static __device__ __forceinline__ v16bf load_frag_a(const __bf16* __restrict__ A,
                                                    int lda, int m0, int kk, int lane) {
    const int m    = m0 + (lane & 15);
    const int koff = (lane & 16) ? 8 : 0;
    const __bf16* p = A + (size_t)m * lda + kk + koff;
    v8bf lo = *(const v8bf*)(p);
    v8bf hi = *(const v8bf*)(p + 16);
    v16bf f;
#pragma unroll
    for (int i = 0; i < 8; ++i) { f[i] = lo[i]; f[i + 8] = hi[i]; }
    return f;
}

static __device__ __forceinline__ v16bf load_frag_b_lds(const __bf16* tile,
                                                        int n, int ks, int lane) {
    const __bf16* p = tile + (n + (lane & 15)) * BPAD + ks + ((lane & 16) ? 8 : 0);
    v8bf lo = *(const v8bf*)(p);       // 16B aligned (rows are 144B)
    v8bf hi = *(const v8bf*)(p + 16);
    v16bf f;
#pragma unroll
    for (int i = 0; i < 8; ++i) { f[i] = lo[i]; f[i + 8] = hi[i]; }
    return f;
}

// ---------------------------------------------------------------------------
// bf16 GEMM, C[M,N](f32) = A[M,K] * B, B given as B^T row-major (BT[n][k]).
// Block = 256 threads = 8 waves; each wave computes 16 x 64; block 128 x 64.
// B^T tile is TDM-DMA'd into a double-buffered LDS tile by wave 0 and shared
// by all 8 waves; A fragments stream from global. All loads for a chunk are
// issued before its 8 WMMAs so waits drain while the matrix pipe runs.
// Optional +addsrc epilogue.
// ---------------------------------------------------------------------------
__global__ __launch_bounds__(256)
void wmma_gemm_bf16(const __bf16* __restrict__ A, const __bf16* __restrict__ BT,
                    float* __restrict__ C, const float* __restrict__ addsrc,
                    int K, int lda, int ldb, int ldc, int Nrows) {
    __shared__ __bf16 ldsB[2][TILE_N * BPAD];

    const int lane = threadIdx.x & 31;
    const int wave = threadIdx.x >> 5;
    const int m0 = blockIdx.x * TILE_M + wave * 16;
    const int n0 = blockIdx.y * TILE_N;

    const __bf16* bt_base = BT + (size_t)n0 * ldb;
    const int nchunks = K / KC;

    if (wave == 0) {
        tdm_load_tile((unsigned)(unsigned long long)&ldsB[0][0],
                      bt_base, (unsigned)ldb, (unsigned)Nrows);
    }

    v8f acc[4] = {v8f{}, v8f{}, v8f{}, v8f{}};

    for (int c = 0; c < nchunks; ++c) {
        // TENSORcnt is already 0 for waves that issued no TDM -> wait is free,
        // and doing it unconditionally avoids an EXEC save/restore sequence.
        __builtin_amdgcn_s_wait_tensorcnt(0);
        __syncthreads();   // LDS buffer (c&1) now valid for all waves
        if (wave == 0 && (c + 1) < nchunks) {
            tdm_load_tile((unsigned)(unsigned long long)&ldsB[(c + 1) & 1][0],
                          bt_base + (size_t)(c + 1) * KC,
                          (unsigned)ldb, (unsigned)Nrows);
        }
        const __bf16* tile = &ldsB[c & 1][0];
        const int kk = c * KC;
        // prefetch next chunk's A rows (speculative; working set is L2-resident)
        if ((c + 1) < nchunks)
            __builtin_prefetch(A + (size_t)(m0 + (lane & 15)) * lda + kk + KC, 0, 1);

        // Issue ALL fragment loads for this chunk up front (2 A from global,
        // 8 B from LDS), then run the 8 WMMAs back-to-back.
        v16bf a0 = load_frag_a(A, lda, m0, kk,      lane);
        v16bf a1 = load_frag_a(A, lda, m0, kk + 32, lane);
        v16bf b0 = load_frag_b_lds(tile,  0,  0, lane);
        v16bf b1 = load_frag_b_lds(tile, 16,  0, lane);
        v16bf b2 = load_frag_b_lds(tile, 32,  0, lane);
        v16bf b3 = load_frag_b_lds(tile, 48,  0, lane);
        v16bf b4 = load_frag_b_lds(tile,  0, 32, lane);
        v16bf b5 = load_frag_b_lds(tile, 16, 32, lane);
        v16bf b6 = load_frag_b_lds(tile, 32, 32, lane);
        v16bf b7 = load_frag_b_lds(tile, 48, 32, lane);

        acc[0] = __builtin_amdgcn_wmma_f32_16x16x32_bf16(false, a0, false, b0, (short)0, acc[0], false, false);
        acc[1] = __builtin_amdgcn_wmma_f32_16x16x32_bf16(false, a0, false, b1, (short)0, acc[1], false, false);
        acc[2] = __builtin_amdgcn_wmma_f32_16x16x32_bf16(false, a0, false, b2, (short)0, acc[2], false, false);
        acc[3] = __builtin_amdgcn_wmma_f32_16x16x32_bf16(false, a0, false, b3, (short)0, acc[3], false, false);
        acc[0] = __builtin_amdgcn_wmma_f32_16x16x32_bf16(false, a1, false, b4, (short)0, acc[0], false, false);
        acc[1] = __builtin_amdgcn_wmma_f32_16x16x32_bf16(false, a1, false, b5, (short)0, acc[1], false, false);
        acc[2] = __builtin_amdgcn_wmma_f32_16x16x32_bf16(false, a1, false, b6, (short)0, acc[2], false, false);
        acc[3] = __builtin_amdgcn_wmma_f32_16x16x32_bf16(false, a1, false, b7, (short)0, acc[3], false, false);
    }

    // C/D layout: VGPR r -> M = m0 + r + ((lane&16)?8:0), N = n0 + (lane&15)
    const int mrow = m0 + ((lane & 16) ? 8 : 0);
    const int ncol = n0 + (lane & 15);
#pragma unroll
    for (int j = 0; j < 4; ++j) {
#pragma unroll
        for (int r = 0; r < 8; ++r) {
            const size_t idx = (size_t)(mrow + r) * ldc + (ncol + j * 16);
            float v = acc[j][r];
            if (addsrc) v += addsrc[idx];
            C[idx] = v;
        }
    }
}

// ---------------------------------------------------------------------------
// Row softmax over k=512 columns: one wave per row, 16 elems per lane.
// ---------------------------------------------------------------------------
__global__ __launch_bounds__(256)
void softmax_rows(const float* __restrict__ logits, int b, int k,
                  __bf16* __restrict__ r_bf, __bf16* __restrict__ rT_bf,
                  float* __restrict__ r_f32) {
    const int lane = threadIdx.x & 31;
    const int row  = blockIdx.x * 8 + (threadIdx.x >> 5);
    if (row >= b) return;
    const float* in = logits + (size_t)row * k;

    float vals[16];
    float mx = -3.402823466e38f;
#pragma unroll
    for (int i = 0; i < 16; ++i) { vals[i] = in[lane + i * 32]; mx = fmaxf(mx, vals[i]); }
#pragma unroll
    for (int off = 16; off > 0; off >>= 1) mx = fmaxf(mx, __shfl_xor(mx, off, 32));

    float sum = 0.f;
#pragma unroll
    for (int i = 0; i < 16; ++i) { vals[i] = __expf(vals[i] - mx); sum += vals[i]; }
#pragma unroll
    for (int off = 16; off > 0; off >>= 1) sum += __shfl_xor(sum, off, 32);

    const float inv = 1.0f / sum;
#pragma unroll
    for (int i = 0; i < 16; ++i) {
        const int col = lane + i * 32;
        const float v = vals[i] * inv;
        r_bf[(size_t)row * k + col] = (__bf16)v;
        if (rT_bf) rT_bf[(size_t)col * b + row] = (__bf16)v;
        if (r_f32) r_f32[(size_t)row * k + col] = v;
    }
}

// ---------------------------------------------------------------------------
// Row L2-normalize p[k,d]: one wave per row; writes bf16 p and bf16 p^T.
// ---------------------------------------------------------------------------
__global__ __launch_bounds__(256)
void normalize_rows(const float* __restrict__ P, int k, int d,
                    __bf16* __restrict__ p_bf, __bf16* __restrict__ pT_bf) {
    const int lane = threadIdx.x & 31;
    const int row  = blockIdx.x * 8 + (threadIdx.x >> 5);
    if (row >= k) return;
    const float* in = P + (size_t)row * d;

    float ss = 0.f;
    for (int i = lane; i < d; i += 32) { const float v = in[i]; ss += v * v; }
#pragma unroll
    for (int off = 16; off > 0; off >>= 1) ss += __shfl_xor(ss, off, 32);

    const float inv = 1.0f / (sqrtf(ss) + 1e-6f);
    for (int i = lane; i < d; i += 32) {
        const float v = in[i] * inv;
        p_bf[(size_t)row * d + i]  = (__bf16)v;
        pT_bf[(size_t)i * k + row] = (__bf16)v;
    }
}

// fp32 -> bf16 elementwise convert
__global__ void cvt_f32_bf16(const float* __restrict__ src, __bf16* __restrict__ dst, int n) {
    const int i = blockIdx.x * blockDim.x + threadIdx.x;
    if (i < n) dst[i] = (__bf16)src[i];
}

// Tiled fp32 -> bf16 transpose through LDS (64x64 tiles, 256 threads)
__global__ __launch_bounds__(256)
void transpose_f32_bf16(const float* __restrict__ src, __bf16* __restrict__ dst,
                        int rows, int cols) {
    __shared__ __bf16 tile[64][65];
    const int r0 = blockIdx.x * 64;
    const int c0 = blockIdx.y * 64;
    const int tx = threadIdx.x & 63;
    const int ty = threadIdx.x >> 6;
#pragma unroll
    for (int p = 0; p < 16; ++p) {
        const int rr = ty + p * 4;
        tile[rr][tx] = (__bf16)src[(size_t)(r0 + rr) * cols + (c0 + tx)];
    }
    __syncthreads();
#pragma unroll
    for (int p = 0; p < 16; ++p) {
        const int rr = ty + p * 4;
        dst[(size_t)(c0 + rr) * rows + (r0 + tx)] = tile[tx][rr];
    }
}

// ---------------------------------------------------------------------------
extern "C" void kernel_launch(void* const* d_in, const int* in_sizes, int n_in,
                              void* d_out, int out_size, void* d_ws, size_t ws_size,
                              hipStream_t stream) {
    const float* cls   = (const float*)d_in[0];   // [b, d] fp32
    const float* proto = (const float*)d_in[1];   // [k, d] fp32, unit rows

    const int d = 1024;
    const int b = in_sizes[0] / d;   // 8192
    const int k = in_sizes[1] / d;   // 512

    // ---- carve workspace ----
    char* ws = (char*)d_ws;
    auto carve = [&](size_t bytes) {
        char* p = ws;
        ws += (bytes + 255) & ~(size_t)255;
        return p;
    };
    __bf16* cls_bf  = (__bf16*)carve((size_t)b * d * 2);   // cls, bf16
    __bf16* clsT_bf = (__bf16*)carve((size_t)d * b * 2);   // cls^T, bf16
    __bf16* p_bf    = (__bf16*)carve((size_t)k * d * 2);   // p, bf16
    __bf16* pT_bf   = (__bf16*)carve((size_t)d * k * 2);   // p^T, bf16
    __bf16* r_bf    = (__bf16*)carve((size_t)b * k * 2);   // r, bf16
    __bf16* rT_bf   = (__bf16*)carve((size_t)k * b * 2);   // r^T, bf16
    float*  logits  = (float*)carve((size_t)b * k * 4);    // cls @ p^T, fp32
    float*  p_f32   = (float*)carve((size_t)k * d * 4);    // r^T @ cls, fp32
    (void)ws_size;

    float* out_r = (float*)d_out;               // [b, k]
    float* out_z = out_r + (size_t)b * k;       // [b, d]

    // ---- init: bf16 copies + cls^T ----
    {
        int n = b * d;
        cvt_f32_bf16<<<(n + 255) / 256, 256, 0, stream>>>(cls, cls_bf, n);
        n = k * d;
        cvt_f32_bf16<<<(n + 255) / 256, 256, 0, stream>>>(proto, p_bf, n);
        transpose_f32_bf16<<<dim3(b / 64, d / 64), 256, 0, stream>>>(cls, clsT_bf, b, d);
    }

    const dim3 gemm_blk(256);
    const dim3 grid_logits(b / TILE_M, k / TILE_N);   // C: [b, k], K = d
    const dim3 grid_pnew(k / TILE_M, d / TILE_N);     // C: [k, d], K = b
    const dim3 grid_z(b / TILE_M, d / TILE_N);        // C: [b, d], K = k

    // ---- 5 EM stages ----
    for (int s = 0; s < 5; ++s) {
        // logits = cls @ p^T   (A = cls_bf [b,d], B^T = p_bf [k,d])
        wmma_gemm_bf16<<<grid_logits, gemm_blk, 0, stream>>>(
            cls_bf, p_bf, logits, nullptr, d, d, d, k, k);
        // r = softmax(logits) -> r_bf, rT_bf
        softmax_rows<<<b / 8, 256, 0, stream>>>(logits, b, k, r_bf, rT_bf, nullptr);
        // p = r^T @ cls        (A = rT_bf [k,b], B^T = clsT_bf [d,b])
        wmma_gemm_bf16<<<grid_pnew, gemm_blk, 0, stream>>>(
            rT_bf, clsT_bf, p_f32, nullptr, b, b, b, d, d);
        // p /= (||p|| + eps) -> p_bf, pT_bf
        normalize_rows<<<k / 8, 256, 0, stream>>>(p_f32, k, d, p_bf, pT_bf);
    }

    // ---- final: r (fp32 out) and z_out = r @ p + cls ----
    wmma_gemm_bf16<<<grid_logits, gemm_blk, 0, stream>>>(
        cls_bf, p_bf, logits, nullptr, d, d, d, k, k);
    softmax_rows<<<b / 8, 256, 0, stream>>>(logits, b, k, r_bf, nullptr, out_r);
    // z = r @ p + cls   (A = r_bf [b,k], B^T = pT_bf [d,k], epilogue +cls)
    wmma_gemm_bf16<<<grid_z, gemm_blk, 0, stream>>>(
        r_bf, pT_bf, out_z, cls, k, k, k, d, d);
}
